// LegendreKANConvNDLayer_75350906241038
// MI455X (gfx1250) — matrix operand order repaired and checked by reading, not compiled, not run on hardware
//
#include <hip/hip_runtime.h>

typedef __attribute__((ext_vector_type(16))) _Float16 v16h;
typedef __attribute__((ext_vector_type(8)))  _Float16 v8h;
typedef __attribute__((ext_vector_type(8)))  float    v8f;

#define NG     4
#define IN_G   16
#define OUT_G  32
#define HW     112
#define NB     32
#define PLANE  (HW*HW)          // 12544
#define EPSF   1e-5f

// workspace layout (bytes)
#define WS_MM  0                // uint minmax[8]  : g*2 = min_enc, g*2+1 = max_enc
#define WS_SUM 64               // float sum[NB*128]
#define WS_SQ  16448            // float sumsq[NB*128]
#define WS_W   32832            // _Float16 weff[288*32*16] = 294912 B (16B aligned)

#define WFRAG_HALVES 512        // one B-fragment set: 32 lanes x 16 halves
#define WG_HALVES    (36*WFRAG_HALVES)   // per-group weight block: 18432 halves = 36864 B

// ---- order-preserving float <-> uint encode for atomic min/max ----
__device__ __forceinline__ unsigned encf(float f) {
    unsigned u = __float_as_uint(f);
    return (u & 0x80000000u) ? ~u : (u | 0x80000000u);
}
__device__ __forceinline__ float decf(unsigned u) {
    unsigned b = (u & 0x80000000u) ? (u & 0x7FFFFFFFu) : ~u;
    return __uint_as_float(b);
}

// ---------------- pass 0a: init workspace ----------------
__global__ void k_init(unsigned* mm, float* sum, float* sq) {
    int i = blockIdx.x * blockDim.x + threadIdx.x;
    if (i < 8) mm[i] = (i & 1) ? 0u : 0xFFFFFFFFu;   // max seed / min seed
    for (int j = i; j < NB * 128; j += gridDim.x * blockDim.x) {
        sum[j] = 0.0f; sq[j] = 0.0f;
    }
}

// ---------------- pass 0b: per-group global min/max ----------------
__global__ void k_minmax(const float* __restrict__ x, unsigned* __restrict__ mm) {
    __shared__ unsigned smn[NG], smx[NG];
    if (threadIdx.x < NG) { smn[threadIdx.x] = 0xFFFFFFFFu; smx[threadIdx.x] = 0u; }
    __syncthreads();
    unsigned lmn[NG] = {0xFFFFFFFFu, 0xFFFFFFFFu, 0xFFFFFFFFu, 0xFFFFFFFFu};
    unsigned lmx[NG] = {0u, 0u, 0u, 0u};
    const long long total = (long long)NB * 64 * PLANE;
    for (long long i = (long long)blockIdx.x * blockDim.x + threadIdx.x; i < total;
         i += (long long)gridDim.x * blockDim.x) {
        int c = (int)((i / PLANE) % 64);
        int g = c >> 4;
        unsigned e = encf(x[i]);
        lmn[g] = min(lmn[g], e);
        lmx[g] = max(lmx[g], e);
    }
    #pragma unroll
    for (int g = 0; g < NG; ++g) {
        atomicMin(&smn[g], lmn[g]);
        atomicMax(&smx[g], lmx[g]);
    }
    __syncthreads();
    if (threadIdx.x < NG) {
        atomicMin(&mm[threadIdx.x * 2 + 0], smn[threadIdx.x]);
        atomicMax(&mm[threadIdx.x * 2 + 1], smx[threadIdx.x]);
    }
}

// ---------------- pass 1: fold base_w into poly_w, swizzle to B-fragment order ----
// b = (((g*9+tap)*2 + chunk)*2 + ntile), one wave per b.
// lane l supplies B fragment element j (0..15): K = (l/16)*16 + j, N = ntile*16 + l%16
__global__ void k_prep(const float* __restrict__ base_w, const float* __restrict__ poly_w,
                       const unsigned* __restrict__ mm, _Float16* __restrict__ weff) {
    int b = blockIdx.x;            // 0..287
    int nt = b & 1;
    int chunk = (b >> 1) & 1;
    int r2 = b >> 2;               // g*9 + tap
    int tap = r2 % 9;
    int g = r2 / 9;
    int l = threadIdx.x;           // 0..31
    float mn = decf(mm[g * 2]), mx = decf(mm[g * 2 + 1]);
    float s = 0.5f * (mx - mn);    // xg = s*P1 + t*P0
    float t = 0.5f * (mx + mn);
    int o = nt * 16 + (l & 15);
    #pragma unroll
    for (int j = 0; j < 16; ++j) {
        int K = ((l >> 4) << 4) + j;
        int c = chunk * 32 + K;    // 0..63 : d*16 + cin
        int d = c >> 4, ci = c & 15;
        float w = poly_w[((size_t)(g * OUT_G + o) * 64 + c) * 9 + tap];
        float bw = base_w[((size_t)(g * OUT_G + o) * IN_G + ci) * 9 + tap];
        if (d == 0)      w += t * bw;
        else if (d == 1) w += s * bw;
        weff[((size_t)b * 32 + l) * 16 + j] = (_Float16)w;
    }
}

// ---------------- pass 2/3: fused basis + implicit-GEMM conv (WMMA) ----------------
// block = 256 threads = 8 waves; tile = 8 rows x 16 cols of one (n,g) image.
// wave wv handles output row h0+wv, 16 cols (M), 32 out channels (2 N-tiles).
// Per-group weights (36 KB) are staged into LDS with ASYNCcnt-tracked
// global_load_async_to_lds_b128, overlapped with the Legendre basis expansion.
__global__ __launch_bounds__(256)
void k_conv(const float* __restrict__ x, const _Float16* __restrict__ weff,
            const unsigned* __restrict__ mm,
            float* __restrict__ sum, float* __restrict__ sq,
            float* __restrict__ out, int phase) {
    // channel-innermost basis tile: [r(10)][cc(18)][ch(64)] f16 = 23040 B
    __shared__ __align__(16) _Float16 basis[10 * 18 * 64];
    // per-group B-fragment weights: 36 fragments x 512 halves = 36864 B
    __shared__ __align__(16) _Float16 wlds[WG_HALVES];

    int z = blockIdx.z;
    int n = z >> 2, g = z & 3;
    int h0 = blockIdx.y * 8, w0 = blockIdx.x * 16;
    int tid = threadIdx.x;

    // ---- kick off async copy of this group's weight block into LDS ----
    // 36864 B = 2304 x b128 ; 256 threads x 9 iterations
    {
        const _Float16* gw = weff + (size_t)g * WG_HALVES;
        #pragma unroll
        for (int it = 0; it < 9; ++it) {
            int idx = (it * 256 + tid) * 8;                       // halves, 16B chunks
            unsigned loff = (unsigned)(unsigned long long)&wlds[idx]; // LDS offset = low 32b of flat addr
            const _Float16* gp = gw + idx;
            asm volatile("global_load_async_to_lds_b128 %0, %1, off"
                         :: "v"(loff), "v"(gp) : "memory");
        }
    }

    float mn = decf(mm[g * 2]), mx = decf(mm[g * 2 + 1]);
    float inv = 2.0f / (mx - mn);

    // ---- stage: load 16ch x 10x18 halo of xg, expand Legendre basis to f16 LDS ----
    for (int slot = tid; slot < IN_G * 10 * 18; slot += 256) {
        int cin = slot / 180;
        int rem = slot % 180;
        int r = rem / 18, cc = rem % 18;
        int row = h0 - 1 + r, col = w0 - 1 + cc;
        bool inb = (row >= 0) & (row < HW) & (col >= 0) & (col < HW);
        float xv = inb ? x[(((size_t)n * 64 + g * IN_G + cin) * HW + row) * HW + col] : 0.0f;
        float xn = inb ? ((xv - mn) * inv - 1.0f) : 0.0f;
        float p0 = inb ? 1.0f : 0.0f;                 // zero-padding applies to P0 too
        float p2 = inb ? (1.5f * xn * xn - 0.5f) : 0.0f;
        float p3 = inb ? (xn * (2.5f * xn * xn - 1.5f)) : 0.0f;
        int base = (r * 18 + cc) * 64 + cin;
        basis[base +  0] = (_Float16)p0;
        basis[base + 16] = (_Float16)xn;
        basis[base + 32] = (_Float16)p2;
        basis[base + 48] = (_Float16)p3;
    }
    asm volatile("s_wait_asynccnt 0x0" ::: "memory");   // weight DMA done (this wave)
    __syncthreads();                                    // + everyone's, + basis visible

    int wv   = tid >> 5;           // output row within tile
    int lane = tid & 31;
    int lm   = lane & 15;          // M index (col) for A; N index (oc) for C
    int hi   = (lane >> 4) * 16;   // K sub-offset for A-fragment upper half-wave

    v8f acc0 = {}; v8f acc1 = {};
    const v8h* wl = (const v8h*)wlds;

    for (int tap = 0; tap < 9; ++tap) {
        int kr = tap / 3, kc = tap % 3;
        int r = wv + kr, cc = lm + kc;
        #pragma unroll
        for (int chunk = 0; chunk < 2; ++chunk) {
            // A fragment: halves 0..15 = ch base..base+15
            const v8h* ap = (const v8h*)&basis[(r * 18 + cc) * 64 + chunk * 32 + hi];
            v8h alo = ap[0];
            v8h ahi = ap[1];
            v16h a;
            #pragma unroll
            for (int i = 0; i < 8; ++i) { a[i] = alo[i]; a[i + 8] = ahi[i]; }

            // B fragments from LDS: 32 contiguous bytes per lane
            int wbase = ((tap * 2 + chunk) * 2) * 64 + lane * 2;   // in v8h units
            v8h b0l = wl[wbase +  0], b0h = wl[wbase +  1];
            v8h b1l = wl[wbase + 64], b1h = wl[wbase + 65];
            v16h b0, b1;
            #pragma unroll
            for (int i = 0; i < 8; ++i) {
                b0[i] = b0l[i]; b0[i + 8] = b0h[i];
                b1[i] = b1l[i]; b1[i + 8] = b1h[i];
            }
            acc0 = __builtin_amdgcn_wmma_f32_16x16x32_f16(false, a, false, b0,
                                                          (short)0, acc0, false, false);
            acc1 = __builtin_amdgcn_wmma_f32_16x16x32_f16(false, a, false, b1,
                                                          (short)0, acc1, false, false);
        }
    }

    // C layout: acc[i] -> pixel M = i + 8*(lane/16), out channel N = lane%16 (+16 for acc1)
    int row   = h0 + wv;
    int mbase = (lane >> 4) * 8;
    int oc0   = g * OUT_G + lm;
    int oc1   = oc0 + 16;

    if (phase == 0) {
        float s0 = 0.f, q0 = 0.f, s1 = 0.f, q1 = 0.f;
        #pragma unroll
        for (int i = 0; i < 8; ++i) {
            float v = acc0[i]; s0 += v; q0 += v * v;
            v = acc1[i];       s1 += v; q1 += v * v;
        }
        atomicAdd(&sum[n * 128 + oc0], s0); atomicAdd(&sq[n * 128 + oc0], q0);
        atomicAdd(&sum[n * 128 + oc1], s1); atomicAdd(&sq[n * 128 + oc1], q1);
    } else {
        const float ip = 1.0f / (float)PLANE;
        float mu0 = sum[n * 128 + oc0] * ip;
        float iv0 = rsqrtf(sq[n * 128 + oc0] * ip - mu0 * mu0 + EPSF);
        float mu1 = sum[n * 128 + oc1] * ip;
        float iv1 = rsqrtf(sq[n * 128 + oc1] * ip - mu1 * mu1 + EPSF);
        #pragma unroll
        for (int i = 0; i < 8; ++i) {
            int col = w0 + mbase + i;
            float yn = (acc0[i] - mu0) * iv0;
            float r0 = yn / (1.0f + __expf(-yn));
            __builtin_nontemporal_store(r0, &out[(((size_t)n * 128 + oc0) * HW + row) * HW + col]);
            yn = (acc1[i] - mu1) * iv1;
            float r1 = yn / (1.0f + __expf(-yn));
            __builtin_nontemporal_store(r1, &out[(((size_t)n * 128 + oc1) * HW + row) * HW + col]);
        }
    }
}

extern "C" void kernel_launch(void* const* d_in, const int* in_sizes, int n_in,
                              void* d_out, int out_size, void* d_ws, size_t ws_size,
                              hipStream_t stream) {
    const float* x      = (const float*)d_in[0];
    const float* base_w = (const float*)d_in[1];
    const float* poly_w = (const float*)d_in[2];
    char* ws = (char*)d_ws;
    unsigned* mm   = (unsigned*)(ws + WS_MM);
    float*    sum  = (float*)(ws + WS_SUM);
    float*    sq   = (float*)(ws + WS_SQ);
    _Float16* weff = (_Float16*)(ws + WS_W);
    float*    out  = (float*)d_out;

    k_init  <<<32,   256, 0, stream>>>(mm, sum, sq);
    k_minmax<<<2048, 256, 0, stream>>>(x, mm);
    k_prep  <<<288,  32,  0, stream>>>(base_w, poly_w, mm, weff);

    dim3 grid(HW / 16, HW / 8, NB * NG);   // (7, 14, 128)
    k_conv<<<grid, 256, 0, stream>>>(x, weff, mm, sum, sq, out, 0);  // stats
    k_conv<<<grid, 256, 0, stream>>>(x, weff, mm, sum, sq, out, 1);  // norm+silu+store
}
